// SSM_38491496907059
// MI455X (gfx1250) — matrix operand order repaired and checked by reading, not compiled, not run on hardware
//
#include <hip/hip_runtime.h>
#include <hip/hip_bf16.h>

// ---------------------------------------------------------------------------
// SSM importance kernel for MI455X (gfx1250, wave32, WMMA + TDM).
//
//   1. xB = x @ B            : bf16 WMMA GEMM (32768x1024x256), f32 accum,
//                              x tiles staged to LDS via TDM (tensor_load_to_lds)
//                              with LDS padding for conflict-free reads.
//   2. scan  -> chunked form : h_final = sum_c [sum_t xb(c,t) A^{63-t}] (A^64)^{63-c}
//      - A powers built by log-depth WMMA GEMMs, then repacked to bf16
//        B-operand fragments so the scan does contiguous vector loads.
//   3. h_proj = h_final @ W^T ; importance = softmax(x . h_proj).
// ---------------------------------------------------------------------------

typedef __attribute__((ext_vector_type(16))) __bf16 v16bf;
typedef __attribute__((ext_vector_type(8)))  float  v8f;
typedef __attribute__((ext_vector_type(4)))  unsigned int u32x4;
typedef __attribute__((ext_vector_type(8)))  int    i32x8;
typedef __attribute__((ext_vector_type(4)))  int    i32x4;

#define D_MODEL   1024
#define STATE_DIM 256
#define BT        8
#define SEQ       4096
#define NROWS     (BT * SEQ)        // 32768
#define CHUNK     64
#define NCHUNK    (SEQ / CHUNK)     // 64
#define MATELEMS  (STATE_DIM * STATE_DIM)  // 65536
#define FRAGLANES 4096              // 8 kb * 16 nt * 32 lanes per matrix

#if defined(__has_builtin)
#if __has_builtin(__builtin_amdgcn_tensor_load_to_lds)
#define HAVE_TDM 1
#endif
#endif
#ifndef HAVE_TDM
#define HAVE_TDM 0
#endif

#if __has_include(<hip/amd_detail/amd_gfx1250_TDM.h>)
#define TDM_6ARG 1
#else
#define TDM_6ARG 0
#endif

__device__ __forceinline__ v8f wmma_bf16(v16bf a, v16bf b, v8f c) {
  return __builtin_amdgcn_wmma_f32_16x16x32_bf16(false, a, false, b, (short)0, c,
                                                 false, false);
}

__device__ __forceinline__ v16bf pack16(float4 q0, float4 q1, float4 q2, float4 q3) {
  v16bf a;
  a[0]=(__bf16)q0.x; a[1]=(__bf16)q0.y; a[2]=(__bf16)q0.z; a[3]=(__bf16)q0.w;
  a[4]=(__bf16)q1.x; a[5]=(__bf16)q1.y; a[6]=(__bf16)q1.z; a[7]=(__bf16)q1.w;
  a[8]=(__bf16)q2.x; a[9]=(__bf16)q2.y; a[10]=(__bf16)q2.z; a[11]=(__bf16)q2.w;
  a[12]=(__bf16)q3.x; a[13]=(__bf16)q3.y; a[14]=(__bf16)q3.z; a[15]=(__bf16)q3.w;
  return a;
}

// A-operand fragment (16x32 bf16) from row-major f32 [ld] per ISA layout.
__device__ __forceinline__ v16bf load_a_frag(const float* __restrict__ src, int ld,
                                             int lane) {
  const float* p = src + (size_t)(lane & 15) * ld;
  int kb = (lane >> 4) << 3;
  float4 q0 = *(const float4*)(p + kb);
  float4 q1 = *(const float4*)(p + kb + 4);
  float4 q2 = *(const float4*)(p + kb + 16);
  float4 q3 = *(const float4*)(p + kb + 20);
  return pack16(q0, q1, q2, q3);
}

// Same but only first 8 rows valid (batch dim padded to 16).
__device__ __forceinline__ v16bf load_a_frag_rows8(const float* __restrict__ src,
                                                   long long bStride, int lane) {
  int r = lane & 15;
  int kb = (lane >> 4) << 3;
  if (r < 8) {
    const float* p = src + (long long)r * bStride;
    float4 q0 = *(const float4*)(p + kb);
    float4 q1 = *(const float4*)(p + kb + 4);
    float4 q2 = *(const float4*)(p + kb + 16);
    float4 q3 = *(const float4*)(p + kb + 20);
    return pack16(q0, q1, q2, q3);
  }
  v16bf z;
#pragma unroll
  for (int e = 0; e < 16; ++e) z[e] = (__bf16)0.0f;
  return z;
}

// B-operand fragment (32x16 bf16) gathered from row-major f32 [ld].
__device__ __forceinline__ v16bf load_b_frag(const float* __restrict__ src, int ld,
                                             int lane) {
  const float* p = src + (lane & 15) + (size_t)((lane >> 4) << 3) * ld;
  v16bf b;
#pragma unroll
  for (int e = 0; e < 16; ++e) b[e] = (__bf16)p[(size_t)(e + (e & 8)) * ld];
  return b;
}

#if HAVE_TDM
// TDM: load a [rows=64] x [cols=32] f32 tile (row stride = D_MODEL elements)
// into LDS with 4-dword padding every 32 dwords (LDS pitch = 36 floats).
__device__ __forceinline__ void tdm_load_x_tile(const float* gsrc, unsigned lds_off) {
  unsigned long long ga = (unsigned long long)(uintptr_t)gsrc;
  u32x4 g0;
  g0[0] = 1u;                                            // count=1 valid descriptor
  g0[1] = lds_off;                                       // lds_addr (bytes)
  g0[2] = (unsigned)ga;                                  // global_addr[31:0]
  g0[3] = (unsigned)((ga >> 32) & 0x01ffffffu) | (2u << 30);  // addr[56:32] | type=2
  i32x8 g1;
  g1[0] = (2 << 16)      // data_size = 4 bytes
        | (1 << 20)      // pad_enable
        | (4 << 22)      // pad_interval: 2^(4+1) = 32 dwords
        | (3 << 25);     // pad_amount: 4 dwords
  g1[1] = 32 << 16;      // tensor_dim0 = 32 (low 16 bits at bit 48)
  g1[2] = 64 << 16;      // tensor_dim0 hi=0 | tensor_dim1 = 64 (low 16 at bit 80)
  g1[3] = 32 << 16;      // tensor_dim1 hi=0 | tile_dim0 = 32
  g1[4] = 64;            // tile_dim1 = 64, tile_dim2 = 0
  g1[5] = D_MODEL;       // tensor_dim0_stride low 32 = 1024
  g1[6] = 0;             // stride hi | tensor_dim1_stride low
  g1[7] = 0;
  i32x4 z4 = {0, 0, 0, 0};
#if TDM_6ARG
  i32x8 z8 = {0, 0, 0, 0, 0, 0, 0, 0};
  __builtin_amdgcn_tensor_load_to_lds(g0, g1, z4, z4, z8, 0);
#else
  __builtin_amdgcn_tensor_load_to_lds(g0, g1, z4, z4, 0);
#endif
}
#endif

// ---------------------------------------------------------------------------
// Kernel 0: repack B [1024,256] f32 -> bf16 B-operand fragments.
// frag[kb(32)][nt(16)][lane(32)][e(16)]
__global__ void ssm_repack_B(const float* __restrict__ B, __bf16* __restrict__ Bfrag) {
  int idx  = blockIdx.x * blockDim.x + threadIdx.x;  // 16384
  int lane = idx & 31;
  int nt   = (idx >> 5) & 15;
  int kb   = idx >> 9;
  int n    = nt * 16 + (lane & 15);
  int k0   = kb * 32 + ((lane >> 4) << 3);
  __bf16* dst = Bfrag + (size_t)idx * 16;
#pragma unroll
  for (int e = 0; e < 16; ++e)
    dst[e] = (__bf16)B[(size_t)(k0 + e + (e & 8)) * STATE_DIM + n];
}

// Kernel 0b: repack nmats power matrices (f32, row-major) into bf16 fragments.
// frag layout per matrix: [kb(8)][nt(16)][lane(32)][e(16)]
__global__ void ssm_repack_mats(const float* __restrict__ mats,
                                __bf16* __restrict__ frag) {
  int m = blockIdx.x;
  const float* src = mats + (size_t)m * MATELEMS;
  for (int base = threadIdx.x; base < FRAGLANES; base += 256) {
    int lane = base & 31;
    int nt   = (base >> 5) & 15;
    int kb   = base >> 9;
    int n    = nt * 16 + (lane & 15);
    int k0   = kb * 32 + ((lane >> 4) << 3);
    __bf16* dst = frag + ((size_t)m * FRAGLANES + base) * 16;
#pragma unroll
    for (int e = 0; e < 16; ++e)
      dst[e] = (__bf16)src[(size_t)(k0 + e + (e & 8)) * STATE_DIM + n];
  }
}

// ---------------------------------------------------------------------------
// Kernel 1: xB = x @ B. 256 threads = 8 waves: wave -> (M-tile 0..3, N-half 0..1).
// x tiles double-buffered in LDS via TDM; B fragments pre-repacked in global.
#define XS_PITCH 36  // 32 dwords + 4 pad dwords (TDM pad) -> conflict-free LDS reads
__global__ void ssm_gemm_xB(const float* __restrict__ x,
                            const __bf16* __restrict__ Bfrag,
                            float* __restrict__ xB) {
  int wave = threadIdx.x >> 5, lane = threadIdx.x & 31;
  int mt = wave >> 1, nHalf = wave & 1;
  int row0 = blockIdx.x * 64;
  v8f acc[8] = {};
  int r = lane & 15;
  int kOff = (lane >> 4) << 3;  // 0 or 8

#if HAVE_TDM
  __shared__ float xs[2][64 * XS_PITCH];  // 18 KB
  if (wave == 0)
    tdm_load_x_tile(x + (size_t)row0 * D_MODEL, (unsigned)(uintptr_t)&xs[0][0]);
  for (int kb = 0; kb < 32; ++kb) {
    __syncthreads();  // previous readers of xs[(kb+1)&1] are done
    if (wave == 0) {
      if (kb + 1 < 32) {
        tdm_load_x_tile(x + (size_t)row0 * D_MODEL + (kb + 1) * 32,
                        (unsigned)(uintptr_t)&xs[(kb + 1) & 1][0]);
        __builtin_amdgcn_s_wait_tensorcnt(1);  // xs[kb&1] complete
      } else {
        __builtin_amdgcn_s_wait_tensorcnt(0);
      }
    }
    __syncthreads();  // xs[kb&1] visible to all waves
    const float* p = &xs[kb & 1][(mt * 16 + r) * XS_PITCH] + kOff;
    float4 q0 = *(const float4*)(p);
    float4 q1 = *(const float4*)(p + 4);
    float4 q2 = *(const float4*)(p + 16);
    float4 q3 = *(const float4*)(p + 20);
    v16bf a = pack16(q0, q1, q2, q3);
#pragma unroll
    for (int j = 0; j < 8; ++j) {
      int nt = nHalf * 8 + j;
      v16bf b = *(const v16bf*)(Bfrag + ((((size_t)kb * 16 + nt) * 32 + lane) << 4));
      acc[j] = wmma_bf16(a, b, acc[j]);
    }
  }
#else
  for (int kb = 0; kb < 32; ++kb) {
    v16bf a = load_a_frag(x + (size_t)(row0 + mt * 16) * D_MODEL + kb * 32,
                          D_MODEL, lane);
#pragma unroll
    for (int j = 0; j < 8; ++j) {
      int nt = nHalf * 8 + j;
      v16bf b = *(const v16bf*)(Bfrag + ((((size_t)kb * 16 + nt) * 32 + lane) << 4));
      acc[j] = wmma_bf16(a, b, acc[j]);
    }
  }
#endif

  int nIn = lane & 15, rb = (lane >> 4) << 3;
#pragma unroll
  for (int j = 0; j < 8; ++j) {
    int n = (nHalf * 8 + j) * 16 + nIn;
    float* d = xB + (size_t)(row0 + mt * 16 + rb) * STATE_DIM + n;
#pragma unroll
    for (int rr = 0; rr < 8; ++rr) d[(size_t)rr * STATE_DIM] = acc[j][rr];
  }
}

// ---------------------------------------------------------------------------
// Kernel 2: identity init for power chains.
__global__ void ssm_init_eye(float* __restrict__ M) {
  int idx = blockIdx.x * blockDim.x + threadIdx.x;  // 65536
  M[idx] = ((idx >> 8) == (idx & 255)) ? 1.0f : 0.0f;
}

// Kernel 3: one doubling level: P[k+i] = P[i] @ P[k], i=1..grid.x
__global__ void ssm_gemm_pow(float* __restrict__ Pw, int k) {
  int i = blockIdx.x + 1;
  const float* X = Pw + (size_t)i * MATELEMS;
  const float* Y = Pw + (size_t)k * MATELEMS;
  float* O = Pw + (size_t)(k + i) * MATELEMS;
  int wave = threadIdx.x >> 5, lane = threadIdx.x & 31;
  int m0 = blockIdx.y * 16, n0 = wave * 16;
  v8f acc = {};
#pragma unroll
  for (int kb = 0; kb < 8; ++kb) {
    v16bf a = load_a_frag(X + (size_t)m0 * STATE_DIM + kb * 32, STATE_DIM, lane);
    v16bf b = load_b_frag(Y + (size_t)(kb * 32) * STATE_DIM + n0, STATE_DIM, lane);
    acc = wmma_bf16(a, b, acc);
  }
  int n = n0 + (lane & 15);
  int rb = m0 + ((lane >> 4) << 3);
#pragma unroll
  for (int rr = 0; rr < 8; ++rr) O[(size_t)(rb + rr) * STATE_DIM + n] = acc[rr];
}

// ---------------------------------------------------------------------------
// Kernel 4: scan contraction  out_c[b,:] = sum_t src(c,t,b,:) @ mats[T-1-t]
// mats supplied as pre-repacked bf16 fragments; block=512 (16 waves, wave=N-tile).
__global__ void ssm_scan_contract(const float* __restrict__ src, long long bStride,
                                  long long tStride, long long chunkStride, int T,
                                  const __bf16* __restrict__ matFrag,
                                  float* __restrict__ out, long long outChunkStride) {
  int c = blockIdx.x;
  const float* s0 = src + (long long)c * chunkStride;
  float* o = out + (long long)c * outChunkStride;
  int wave = threadIdx.x >> 5, lane = threadIdx.x & 31;
  int n0 = wave << 4;
  v8f acc = {};
  for (int t = 0; t < T; ++t) {
    const __bf16* mf =
        matFrag + (((size_t)(T - 1 - t) * FRAGLANES + (size_t)wave * 32 + lane) << 4);
    const float* st = s0 + (long long)t * tStride;
#pragma unroll
    for (int kb = 0; kb < 8; ++kb) {
      v16bf a = load_a_frag_rows8(st + kb * 32, bStride, lane);
      v16bf b = *(const v16bf*)(mf + ((size_t)kb << 13));  // kb*512 frag-lanes *16
      acc = wmma_bf16(a, b, acc);
    }
  }
  if (lane < 16) {  // rows 0..7 (valid batches) live in lanes 0..15
    int n = n0 + lane;
#pragma unroll
    for (int rr = 0; rr < 8; ++rr) o[(size_t)rr * STATE_DIM + n] = acc[rr];
  }
}

// ---------------------------------------------------------------------------
// Kernel 5: h_proj[b,d] = sum_n h_final[b,n] * W[d,n]
__global__ void ssm_hproj(const float* __restrict__ hf, const float* __restrict__ W,
                          float* __restrict__ hp) {
  int idx = blockIdx.x * blockDim.x + threadIdx.x;  // 8192
  int b = idx >> 10, d = idx & 1023;
  const float* h = hf + (size_t)b * STATE_DIM;
  const float* w = W + (size_t)d * STATE_DIM;
  float s = 0.f;
#pragma unroll 4
  for (int n = 0; n < STATE_DIM; ++n) s += h[n] * w[n];
  hp[idx] = s;
}

// Kernel 6: raw[b,s] = dot(x[b,s,:], hp[b,:]) (one wave per row, wave32 reduce)
__global__ void ssm_dotx(const float* __restrict__ x, const float* __restrict__ hp,
                         float* __restrict__ out) {
  int row = blockIdx.x * 8 + (threadIdx.x >> 5);
  int lane = threadIdx.x & 31;
  int b = row >> 12;
  const float* xr = x + (size_t)row * D_MODEL;
  const float* h = hp + (size_t)b * D_MODEL;
  float s = 0.f;
#pragma unroll 4
  for (int i = lane; i < D_MODEL; i += 32) s += xr[i] * h[i];
#pragma unroll
  for (int off = 16; off > 0; off >>= 1) s += __shfl_down(s, off, 32);
  if (lane == 0) out[row] = s;
}

// Kernel 7: in-place softmax over each 4096-row.
__global__ void ssm_softmax(float* __restrict__ out) {
  float* row = out + (size_t)blockIdx.x * SEQ;
  int tid = threadIdx.x, lane = tid & 31, wave = tid >> 5;
  __shared__ float sh[8];
  float v[16];
  float mx = -3.402823466e38f;
#pragma unroll
  for (int i = 0; i < 16; ++i) { v[i] = row[tid + (i << 8)]; mx = fmaxf(mx, v[i]); }
#pragma unroll
  for (int off = 16; off > 0; off >>= 1) mx = fmaxf(mx, __shfl_down(mx, off, 32));
  if (lane == 0) sh[wave] = mx;
  __syncthreads();
  mx = sh[0];
#pragma unroll
  for (int w = 1; w < 8; ++w) mx = fmaxf(mx, sh[w]);
  __syncthreads();
  float sum = 0.f;
#pragma unroll
  for (int i = 0; i < 16; ++i) { v[i] = __expf(v[i] - mx); sum += v[i]; }
#pragma unroll
  for (int off = 16; off > 0; off >>= 1) sum += __shfl_down(sum, off, 32);
  if (lane == 0) sh[wave] = sum;
  __syncthreads();
  sum = 0.f;
#pragma unroll
  for (int w = 0; w < 8; ++w) sum += sh[w];
  float inv = 1.0f / sum;
#pragma unroll
  for (int i = 0; i < 16; ++i) row[tid + (i << 8)] = v[i] * inv;
}

// ---------------------------------------------------------------------------
extern "C" void kernel_launch(void* const* d_in, const int* in_sizes, int n_in,
                              void* d_out, int out_size, void* d_ws, size_t ws_size,
                              hipStream_t stream) {
  const float* x = (const float*)d_in[0];
  const float* A = (const float*)d_in[1];
  const float* B = (const float*)d_in[2];
  const float* W = (const float*)d_in[3];
  float* out = (float*)d_out;  // [8*4096 importance][8*256 h_final]

  // Workspace carve-up
  char* ws = (char*)d_ws;
  __bf16* Bfrag = (__bf16*)ws;                                   // 512 KB
  float* xB = (float*)(ws + (size_t)(512 << 10));                // 32 MB
  float* P  = xB + (size_t)NROWS * STATE_DIM;                    // 65 * 256KB
  float* Q  = P + (size_t)65 * MATELEMS;                         // 64 * 256KB
  float* hc = Q + (size_t)64 * MATELEMS;                         // 64*8*256
  float* hp = hc + (size_t)NCHUNK * BT * STATE_DIM;              // 8*1024
  __bf16* Pfrag = (__bf16*)(hp + BT * D_MODEL);                  // 8 MB
  __bf16* Qfrag = Pfrag + (size_t)64 * FRAGLANES * 16;           // 8 MB

  // 1. Repack B and run the big projection GEMM (TDM-staged x tiles).
  ssm_repack_B<<<64, 256, 0, stream>>>(B, Bfrag);
  ssm_gemm_xB<<<NROWS / 64, 256, 0, stream>>>(x, Bfrag, xB);

  // 2. Matrix powers P[j] = A^j, j=0..64 (log-depth doubling).
  ssm_init_eye<<<256, 256, 0, stream>>>(P);
  hipMemcpyAsync(P + MATELEMS, A, (size_t)MATELEMS * sizeof(float),
                 hipMemcpyDeviceToDevice, stream);
  for (int k = 1; k <= 32; k <<= 1) {
    int imax = (k < 64 - k) ? k : (64 - k);
    ssm_gemm_pow<<<dim3(imax, 16), 512, 0, stream>>>(P, k);
  }
  // Q[j] = (A^64)^j, j=0..63
  ssm_init_eye<<<256, 256, 0, stream>>>(Q);
  hipMemcpyAsync(Q + MATELEMS, P + (size_t)64 * MATELEMS,
                 (size_t)MATELEMS * sizeof(float), hipMemcpyDeviceToDevice, stream);
  for (int k = 1; k <= 32; k <<= 1) {
    int imax = (k < 63 - k) ? k : (63 - k);
    ssm_gemm_pow<<<dim3(imax, 16), 512, 0, stream>>>(Q, k);
  }
  // Repack powers into bf16 fragments for the scan contractions.
  ssm_repack_mats<<<64, 256, 0, stream>>>(P, Pfrag);
  ssm_repack_mats<<<64, 256, 0, stream>>>(Q, Qfrag);

  // 3. Per-chunk scans:  hc[c][b,:] = sum_t xB[b, c*64+t, :] @ A^{63-t}
  ssm_scan_contract<<<NCHUNK, 512, 0, stream>>>(
      xB, (long long)SEQ * STATE_DIM, (long long)STATE_DIM,
      (long long)CHUNK * STATE_DIM, CHUNK, Pfrag, hc, (long long)BT * STATE_DIM);
  // 4. Final combine: h_final = sum_c hc[c] @ (A^64)^{63-c} -> out tail
  ssm_scan_contract<<<1, 512, 0, stream>>>(
      hc, (long long)STATE_DIM, (long long)BT * STATE_DIM, 0LL, NCHUNK, Qfrag,
      out + NROWS, 0LL);

  // 5. Output projections + softmax.
  ssm_hproj<<<32, 256, 0, stream>>>(out + NROWS, W, hp);
  ssm_dotx<<<NROWS / 8, 256, 0, stream>>>(x, hp, out);
  ssm_softmax<<<BT, 256, 0, stream>>>(out);
}
